// MiniformerModule_37546604101745
// MI455X (gfx1250) — compile-verified, matrix operand order checked by reading, not run on hardware
//
#include <hip/hip_runtime.h>

typedef unsigned short u16;
typedef unsigned int   u32;
typedef __attribute__((ext_vector_type(16))) __bf16 v16bf;
typedef __attribute__((ext_vector_type(16))) short  v16s;
typedef __attribute__((ext_vector_type(8)))  float  v8f;

#define DEVFN static __device__ __forceinline__

static constexpr int N  = 512;
static constexpr int S  = 384;
static constexpr int ZC = 128;
static constexpr int H  = 16;
static constexpr int DH = 24;   // S / H
static constexpr int DP = 32;   // padded head dim

// ---------------- helpers ----------------
DEVFN u16 f2bf(float f) { return __builtin_bit_cast(u16, (__bf16)f); }  // native cvt
DEVFN float bf2f(u16 h) { return __builtin_bit_cast(float, (u32)h << 16); }
DEVFN float sigmoidf(float x) { return 1.f / (1.f + __expf(-x)); }
DEVFN v8f zero8() { v8f z = {0.f,0.f,0.f,0.f,0.f,0.f,0.f,0.f}; return z; }

DEVFN v8f wmma_bf16(v16s a, v16s b, v8f c) {
  return __builtin_amdgcn_wmma_f32_16x16x32_bf16(
      false, __builtin_bit_cast(v16bf, a),
      false, __builtin_bit_cast(v16bf, b),
      (short)0, c, false, false);
}

union FragPack { v16s v; uint4 u[2]; };
union Pack8 { uint4 u; u16 h[8]; };

// Fragment load, table-order k layout (k = 8h+s for s<8, 16+8h+s-8 for s>=8). Works for both A (rows=M) and transposed-B
// (rows=N). Element (r, k) at p[r*rs + k], rs multiple of 8, p 16B-aligned.
DEVFN v16s frag_ld(const u16* p, int rs, int lane) {
  int l = lane & 15, hi = lane >> 4;
  const u16* q = p + l * rs + hi * 8;
  FragPack r;
  r.u[0] = *(const uint4*)(q);
  r.u[1] = *(const uint4*)(q + 16);
  return r.v;
}

DEVFN float loadF(const float* p) { return *p; }
DEVFN float loadF(const u16* p)   { return bf2f(*p); }

// ---------------- f32 -> bf16 transpose conversion: src[K][NC] -> dst[NC][K] ----------------
__global__ void k_cvt_t(const float* __restrict__ src, u16* __restrict__ dst, int K, int NC) {
  int k = blockIdx.x * 16 + (threadIdx.x & 15);
  int n = blockIdx.y * 16 + (threadIdx.x >> 4);
  dst[(long)n * K + k] = f2bf(src[(long)k * NC + n]);
}

// ---------------- LayerNorm over last dim -> bf16 ----------------
template <typename T, int D>
__global__ void k_ln(const T* x, const float* __restrict__ g,
                     const float* __restrict__ bt, u16* out, long M) {
  long row = (long)blockIdx.x * 8 + (threadIdx.x >> 5);
  int lane = threadIdx.x & 31;
  if (row >= M) return;
  const T* xr = x + row * (long)D;
  float vals[D / 32];
  float s = 0.f;
#pragma unroll
  for (int j = 0; j < D / 32; ++j) { vals[j] = loadF(xr + lane + j * 32); s += vals[j]; }
#pragma unroll
  for (int o = 16; o >= 1; o >>= 1) s += __shfl_xor(s, o, 32);
  float mean = s * (1.f / D);
  float var = 0.f;
#pragma unroll
  for (int j = 0; j < D / 32; ++j) { float d = vals[j] - mean; var += d * d; }
#pragma unroll
  for (int o = 16; o >= 1; o >>= 1) var += __shfl_xor(var, o, 32);
  float rstd = rsqrtf(var * (1.f / D) + 1e-5f);
  u16* orow = out + row * (long)D;
#pragma unroll
  for (int j = 0; j < D / 32; ++j) {
    int c = lane + j * 32;
    orow[c] = f2bf((vals[j] - mean) * rstd * g[c] + bt[c]);
  }
}

// ---------------- dual 128x128 GEMM on 64-row tiles: sigmoid(A1@W1) * (A2@W2) ----------------
// CM=1: * pair_mask, write bf16 channel-major out[col][row] (packed 16B stores)
// CM=0: += into f32 resid (pixel-major)
template <int PM, int CM>
__global__ void k_dual128(const u16* __restrict__ A1, const u16* __restrict__ W1T,
                          const u16* __restrict__ A2, const u16* __restrict__ W2T,
                          const float* __restrict__ pm, float* __restrict__ resid,
                          u16* __restrict__ outb, long NNtot) {
  __shared__ __align__(16) u16 sA1[64 * 128];
  __shared__ __align__(16) u16 sA2[64 * 128];
  long row0 = (long)blockIdx.x * 64;
  for (int t = threadIdx.x; t < 1024; t += 256) {
    ((uint4*)sA1)[t] = ((const uint4*)(A1 + row0 * 128))[t];
    ((uint4*)sA2)[t] = ((const uint4*)(A2 + row0 * 128))[t];
  }
  __syncthreads();
  int wave = threadIdx.x >> 5, lane = threadIdx.x & 31;
  int l = lane & 15, hi = lane >> 4;
  int n0 = wave * 16;
  v8f ga[4], ta[4];
#pragma unroll
  for (int rg = 0; rg < 4; ++rg) { ga[rg] = zero8(); ta[rg] = zero8(); }
#pragma unroll
  for (int kk = 0; kk < 4; ++kk) {
    v16s fb1 = frag_ld(W1T + n0 * 128 + kk * 32, 128, lane);
    v16s fb2 = frag_ld(W2T + n0 * 128 + kk * 32, 128, lane);
#pragma unroll
    for (int rg = 0; rg < 4; ++rg) {
      v16s fa1 = frag_ld(sA1 + rg * 16 * 128 + kk * 32, 128, lane);
      v16s fa2 = frag_ld(sA2 + rg * 16 * 128 + kk * 32, 128, lane);
      ga[rg] = wmma_bf16(fa1, fb1, ga[rg]);
      ta[rg] = wmma_bf16(fa2, fb2, ta[rg]);
    }
  }
#pragma unroll
  for (int rg = 0; rg < 4; ++rg) {
    long rbase = row0 + rg * 16 + hi * 8;   // 8 consecutive rows per lane
    if (CM) {
      float pmv[8];
      if (PM) {
        float4 p0 = *(const float4*)(pm + rbase);
        float4 p1 = *(const float4*)(pm + rbase + 4);
        pmv[0]=p0.x; pmv[1]=p0.y; pmv[2]=p0.z; pmv[3]=p0.w;
        pmv[4]=p1.x; pmv[5]=p1.y; pmv[6]=p1.z; pmv[7]=p1.w;
      }
      Pack8 pk;
#pragma unroll
      for (int v = 0; v < 8; ++v) {
        float val = sigmoidf(ga[rg][v]) * ta[rg][v];
        if (PM) val *= pmv[v];
        pk.h[v] = f2bf(val);
      }
      *(uint4*)(outb + (size_t)(n0 + l) * NNtot + rbase) = pk.u;
    } else {
#pragma unroll
      for (int v = 0; v < 8; ++v) {
        float val = sigmoidf(ga[rg][v]) * ta[rg][v];
        resid[(rbase + v) * 128 + n0 + l] += val;
      }
    }
  }
}

// ---------------- triangle einsum: x[i,j,c] = sum_k a[i,k,c]*b[j,k,c] ----------------
// a,b channel-major [c][i][k] bf16; 64x32 (i,j) tile x 8 channels per block.
// Operands are L2-resident (67MB each vs 192MB L2); output staged in LDS and
// flushed as contiguous 16B (8-channel) stores.
__global__ void k_tri_einsum(const u16* __restrict__ a, const u16* __restrict__ b,
                             u16* __restrict__ x) {
  __shared__ __align__(16) u16 sO[64 * 32 * 8];   // [i][j][c]
  int i0 = blockIdx.x * 64, j0 = blockIdx.y * 32, c0 = blockIdx.z * 8;
  int wave = threadIdx.x >> 5, lane = threadIdx.x & 31;
  int ch = c0 + wave;
  int l = lane & 15, hi = lane >> 4;
  const u16* aP = a + (size_t)ch * N * N;
  const u16* bP = b + (size_t)ch * N * N;
  v8f acc[4][2];
#pragma unroll
  for (int rg = 0; rg < 4; ++rg) { acc[rg][0] = zero8(); acc[rg][1] = zero8(); }
  for (int kt = 0; kt < 16; ++kt) {
    int k0 = kt * 32;
    v16s fb0 = frag_ld(bP + (j0) * N + k0, N, lane);
    v16s fb1 = frag_ld(bP + (j0 + 16) * N + k0, N, lane);
#pragma unroll
    for (int rg = 0; rg < 4; ++rg) {
      v16s fa = frag_ld(aP + (i0 + rg * 16) * N + k0, N, lane);
      acc[rg][0] = wmma_bf16(fa, fb0, acc[rg][0]);
      acc[rg][1] = wmma_bf16(fa, fb1, acc[rg][1]);
    }
  }
#pragma unroll
  for (int rg = 0; rg < 4; ++rg)
#pragma unroll
    for (int cg = 0; cg < 2; ++cg)
#pragma unroll
      for (int v = 0; v < 8; ++v) {
        int il = rg * 16 + v + hi * 8;
        int jl = cg * 16 + l;
        sO[(il * 32 + jl) * 8 + wave] = f2bf(acc[rg][cg][v]);
      }
  __syncthreads();
  for (int t = threadIdx.x; t < 2048; t += 256) {
    int il = t >> 5, jl = t & 31;
    *(uint4*)(x + ((long)(i0 + il) * N + j0 + jl) * ZC + c0) = ((const uint4*)sO)[t];
  }
}

// ---------------- fused SwiGLU transition (32-row tiles): y += (silu(LN@W1)*(LN@W2))@W3 ----------------
// W1T,W2T: [H4][D] (transposed); W3T: [D][H4] (transposed)
template <int D, int H4>
__global__ void k_transition(float* y, const float* __restrict__ g, const float* __restrict__ b,
                             const u16* __restrict__ W1T, const u16* __restrict__ W2T,
                             const u16* __restrict__ W3T, long M) {
  constexpr int CTPW = D / 128;
  __shared__ __align__(16) u16 sX[32 * D];
  __shared__ __align__(16) u16 sH[32 * 128];
  int wave = threadIdx.x >> 5, lane = threadIdx.x & 31;
  long row0 = (long)blockIdx.x * 32;
  {  // LayerNorm: 4 rows per wave (2 x 16-lane groups)
    int l16 = lane & 15;
#pragma unroll
    for (int rr = 0; rr < 2; ++rr) {
      int r = wave * 4 + rr * 2 + (lane >> 4);
      const float* xr = y + (row0 + r) * D;
      float vals[D / 16];
      float s = 0.f;
#pragma unroll
      for (int j = 0; j < D / 16; ++j) { vals[j] = xr[l16 + j * 16]; s += vals[j]; }
#pragma unroll
      for (int o = 8; o >= 1; o >>= 1) s += __shfl_xor(s, o, 32);
      float mean = s * (1.f / D);
      float var = 0.f;
#pragma unroll
      for (int j = 0; j < D / 16; ++j) { float d = vals[j] - mean; var += d * d; }
#pragma unroll
      for (int o = 8; o >= 1; o >>= 1) var += __shfl_xor(var, o, 32);
      float rstd = rsqrtf(var * (1.f / D) + 1e-5f);
#pragma unroll
      for (int j = 0; j < D / 16; ++j) {
        int c = l16 + j * 16;
        sX[r * D + c] = f2bf((vals[j] - mean) * rstd * g[c] + b[c]);
      }
    }
  }
  __syncthreads();
  int l = lane & 15, hi = lane >> 4;
  v8f acc[2][CTPW];
#pragma unroll
  for (int rg = 0; rg < 2; ++rg)
#pragma unroll
    for (int t = 0; t < CTPW; ++t) acc[rg][t] = zero8();
  for (int hc = 0; hc < H4 / 128; ++hc) {
    int n0h = hc * 128 + wave * 16;
    v8f a10 = zero8(), a11 = zero8(), a20 = zero8(), a21 = zero8();
#pragma unroll
    for (int kk = 0; kk < D / 32; ++kk) {
      v16s fa0 = frag_ld(sX + kk * 32, D, lane);
      v16s fa1 = frag_ld(sX + 16 * D + kk * 32, D, lane);
      v16s fb1 = frag_ld(W1T + (long)n0h * D + kk * 32, D, lane);
      v16s fb2 = frag_ld(W2T + (long)n0h * D + kk * 32, D, lane);
      a10 = wmma_bf16(fa0, fb1, a10);
      a11 = wmma_bf16(fa1, fb1, a11);
      a20 = wmma_bf16(fa0, fb2, a20);
      a21 = wmma_bf16(fa1, fb2, a21);
    }
#pragma unroll
    for (int rg = 0; rg < 2; ++rg) {
      v8f h1 = rg ? a11 : a10;
      v8f h2 = rg ? a21 : a20;
#pragma unroll
      for (int v = 0; v < 8; ++v) {
        float x1 = h1[v];
        float hval = (x1 * sigmoidf(x1)) * h2[v];
        sH[(rg * 16 + v + hi * 8) * 128 + wave * 16 + l] = f2bf(hval);
      }
    }
    __syncthreads();
#pragma unroll
    for (int t = 0; t < CTPW; ++t) {
      int n0 = (wave + t * 8) * 16;
#pragma unroll
      for (int kk = 0; kk < 4; ++kk) {
        v16s fb = frag_ld(W3T + (long)n0 * H4 + hc * 128 + kk * 32, H4, lane);
        v16s fh0 = frag_ld(sH + kk * 32, 128, lane);
        v16s fh1 = frag_ld(sH + 16 * 128 + kk * 32, 128, lane);
        acc[0][t] = wmma_bf16(fh0, fb, acc[0][t]);
        acc[1][t] = wmma_bf16(fh1, fb, acc[1][t]);
      }
    }
    __syncthreads();
  }
#pragma unroll
  for (int rg = 0; rg < 2; ++rg)
#pragma unroll
    for (int t = 0; t < CTPW; ++t)
#pragma unroll
      for (int v = 0; v < 8; ++v)
        y[(row0 + rg * 16 + v + hi * 8) * D + (wave + t * 8) * 16 + l] += acc[rg][t][v];
}

// ---------------- generic GEMM, NC = TPW*128 cols; WT is [NC][K] ----------------
// OMODE: 0 f32 plain | 1 bf16 plain | 2 f32 residual add
//        3 bf16 head-padded [row][h*32+d] | 4 bf16 head-padded transposed [h*32+d][row]
template <int TPW, int ACT, int BIAS, int MUL, int OMODE>
__global__ void k_gemm(const u16* __restrict__ A, const u16* __restrict__ WT,
                       const float* __restrict__ bias, const float* __restrict__ mul,
                       float* outf, u16* outb, int K, long M) {
  constexpr int NC = TPW * 128;
  __shared__ __align__(16) u16 sA[32 * 384];
  long row0 = (long)blockIdx.x * 32;
  int nv4 = 32 * K / 8;
  for (int t = threadIdx.x; t < nv4; t += 256)
    ((uint4*)sA)[t] = ((const uint4*)(A + row0 * K))[t];
  __syncthreads();
  int wave = threadIdx.x >> 5, lane = threadIdx.x & 31;
  int l = lane & 15, hi = lane >> 4;
  v8f acc[2][TPW];
#pragma unroll
  for (int rg = 0; rg < 2; ++rg)
#pragma unroll
    for (int t = 0; t < TPW; ++t) acc[rg][t] = zero8();
  for (int kk = 0; kk < K / 32; ++kk) {
    v16s fa0 = frag_ld(sA + kk * 32, K, lane);
    v16s fa1 = frag_ld(sA + 16 * K + kk * 32, K, lane);
#pragma unroll
    for (int t = 0; t < TPW; ++t) {
      int n0 = (wave + t * 8) * 16;
      v16s fb = frag_ld(WT + (long)n0 * K + kk * 32, K, lane);
      acc[0][t] = wmma_bf16(fa0, fb, acc[0][t]);
      acc[1][t] = wmma_bf16(fa1, fb, acc[1][t]);
    }
  }
#pragma unroll
  for (int rg = 0; rg < 2; ++rg)
#pragma unroll
    for (int t = 0; t < TPW; ++t) {
      int col = (wave + t * 8) * 16 + l;
      long rbase = row0 + rg * 16 + hi * 8;
      float vals[8];
#pragma unroll
      for (int v = 0; v < 8; ++v) {
        float val = acc[rg][t][v];
        if (BIAS) val += bias[col];
        if (ACT == 1) val = sigmoidf(val);
        if (MUL) val *= mul[(rbase + v) * NC + col];
        vals[v] = val;
      }
      if (OMODE == 0) {
#pragma unroll
        for (int v = 0; v < 8; ++v) outf[(rbase + v) * NC + col] = vals[v];
      } else if (OMODE == 1) {
#pragma unroll
        for (int v = 0; v < 8; ++v) outb[(rbase + v) * NC + col] = f2bf(vals[v]);
      } else if (OMODE == 2) {
#pragma unroll
        for (int v = 0; v < 8; ++v) outf[(rbase + v) * NC + col] += vals[v];
      } else if (OMODE == 3) {
        int hh = col / DH, dd = col - hh * DH;
#pragma unroll
        for (int v = 0; v < 8; ++v) outb[(rbase + v) * (H * DP) + hh * DP + dd] = f2bf(vals[v]);
      } else {  // OMODE == 4: transposed padded (V): packed 16B store
        int hh = col / DH, dd = col - hh * DH;
        Pack8 pk;
#pragma unroll
        for (int v = 0; v < 8; ++v) pk.h[v] = f2bf(vals[v]);
        *(uint4*)(outb + (size_t)(hh * DP + dd) * N + rbase) = pk.u;
      }
    }
}

// ---------------- pair-bias projection: [M,128] @ [128,16] -> f32; WT is [16][128] ----------------
__global__ void k_gemm_nc16(const u16* __restrict__ A, const u16* __restrict__ WT,
                            float* __restrict__ out, long M) {
  __shared__ __align__(16) u16 sA[128 * 128];
  long row0 = (long)blockIdx.x * 128;
  for (int t = threadIdx.x; t < 2048; t += 256)
    ((uint4*)sA)[t] = ((const uint4*)(A + row0 * 128))[t];
  __syncthreads();
  int wave = threadIdx.x >> 5, lane = threadIdx.x & 31;
  int l = lane & 15, hi = lane >> 4;
  v8f acc = zero8();
#pragma unroll
  for (int kk = 0; kk < 4; ++kk) {
    v16s fa = frag_ld(sA + wave * 16 * 128 + kk * 32, 128, lane);
    v16s fb = frag_ld(WT + kk * 32, 128, lane);
    acc = wmma_bf16(fa, fb, acc);
  }
#pragma unroll
  for (int v = 0; v < 8; ++v)
    out[(row0 + wave * 16 + v + hi * 8) * 16 + l] = acc[v];
}

// ---------------- attention core: one wave per (head, 16 query rows) ----------------
// qp,kp: [N][H*32] bf16 zero-padded; vt: [H*32][N] bf16 zero-padded rows
__global__ void k_attn(const u16* __restrict__ qp, const u16* __restrict__ kp,
                       const u16* __restrict__ vt, const float* __restrict__ biasz,
                       const float* __restrict__ mask, float* __restrict__ o) {
  __shared__ float sS[16 * 512];
  int task = blockIdx.x;
  int h = task >> 5;          // N/16 == 32 tasks per head
  int q0 = (task & 31) * 16;
  int lane = threadIdx.x & 31;
  int l = lane & 15, hi = lane >> 4;
  const float sc = 0.2041241452f;  // 1/sqrt(24)
  v16s fq = frag_ld(qp + (long)q0 * (H * DP) + h * DP, H * DP, lane);
  for (int jt = 0; jt < 32; ++jt) {
    int j0 = jt * 16;
    v16s fk = frag_ld(kp + (long)j0 * (H * DP) + h * DP, H * DP, lane);
    v8f s8 = zero8();
    s8 = wmma_bf16(fq, fk, s8);
#pragma unroll
    for (int v = 0; v < 8; ++v) {
      int m = v + hi * 8, jj = j0 + l;
      float val = s8[v] * sc + biasz[((long)(q0 + m) * N + jj) * H + h]
                + (1.f - mask[jj]) * (-1e9f);
      sS[m * 512 + jj] = val;
    }
  }
  // softmax rows
  for (int m = 0; m < 16; ++m) {
    float mx = -1e30f;
    for (int j = lane; j < 512; j += 32) mx = fmaxf(mx, sS[m * 512 + j]);
#pragma unroll
    for (int ofs = 16; ofs >= 1; ofs >>= 1) mx = fmaxf(mx, __shfl_xor(mx, ofs, 32));
    float sum = 0.f;
    for (int j = lane; j < 512; j += 32) {
      float e = __expf(sS[m * 512 + j] - mx);
      sS[m * 512 + j] = e;
      sum += e;
    }
#pragma unroll
    for (int ofs = 16; ofs >= 1; ofs >>= 1) sum += __shfl_xor(sum, ofs, 32);
    float inv = 1.f / sum;
    for (int j = lane; j < 512; j += 32) sS[m * 512 + j] *= inv;
  }
  // o = attn @ v  (two d-col tiles; second tile rows 24..31 are zero-padded in vt)
  v8f acc0 = zero8(), acc1 = zero8();
  for (int kt = 0; kt < 16; ++kt) {
    int k0 = kt * 32;
    const float* ps = &sS[l * 512 + k0 + hi * 8];
    float4 w0 = *(const float4*)(ps);
    float4 w1 = *(const float4*)(ps + 4);
    float4 w2 = *(const float4*)(ps + 16);
    float4 w3 = *(const float4*)(ps + 20);
    v16s fa;
    fa[0]=(short)f2bf(w0.x); fa[1]=(short)f2bf(w0.y); fa[2]=(short)f2bf(w0.z); fa[3]=(short)f2bf(w0.w);
    fa[4]=(short)f2bf(w1.x); fa[5]=(short)f2bf(w1.y); fa[6]=(short)f2bf(w1.z); fa[7]=(short)f2bf(w1.w);
    fa[8]=(short)f2bf(w2.x); fa[9]=(short)f2bf(w2.y); fa[10]=(short)f2bf(w2.z); fa[11]=(short)f2bf(w2.w);
    fa[12]=(short)f2bf(w3.x); fa[13]=(short)f2bf(w3.y); fa[14]=(short)f2bf(w3.z); fa[15]=(short)f2bf(w3.w);
    v16s fv0 = frag_ld(vt + (long)(h * DP) * N + k0, N, lane);
    v16s fv1 = frag_ld(vt + (long)(h * DP + 16) * N + k0, N, lane);
    acc0 = wmma_bf16(fa, fv0, acc0);
    acc1 = wmma_bf16(fa, fv1, acc1);
  }
#pragma unroll
  for (int v = 0; v < 8; ++v) {
    int m = v + hi * 8;
    o[(long)(q0 + m) * S + h * DH + l] = acc0[v];
    if (l < DH - 16) o[(long)(q0 + m) * S + h * DH + 16 + l] = acc1[v];
  }
}

// ---------------- host ----------------
extern "C" void kernel_launch(void* const* d_in, const int* in_sizes, int n_in,
                              void* d_out, int out_size, void* d_ws, size_t ws_size,
                              hipStream_t stream) {
  (void)n_in; (void)out_size; (void)ws_size;
  const float* s_in  = (const float*)d_in[0];
  const float* z_in  = (const float*)d_in[1];
  const float* mask  = (const float*)d_in[2];
  const float* pmask = (const float*)d_in[3];
  enum { P_GS = 4, P_BS, P_WQ, P_BQ, P_WK, P_WV, P_GZB, P_BZB, P_WZB, P_WG, P_WO,
         P_TGIN, P_TBIN, P_WA, P_WAG, P_WB, P_WBG, P_TGOUT, P_TBOUT, P_WOG, P_WOT,
         P_TSG, P_TSB, P_WS1, P_WS2, P_WS3, P_TZG, P_TZB, P_WZ1, P_WZ2, P_WZ3 };
  auto PF = [&](int idx, int layer) -> const float* {
    long per = (long)in_sizes[idx] / 2;
    return (const float*)d_in[idx] + (long)layer * per;
  };

  float* s_w = (float*)d_out;                       // [N,S]
  float* z_w = (float*)d_out + (long)N * S;         // [N,N,ZC]

  char* ws = (char*)d_ws;
  size_t off = 0;
  auto alloc = [&](size_t bytes) -> void* {
    off = (off + 255) & ~(size_t)255;
    void* p = ws + off;
    off += bytes;
    return p;
  };
  const long NN = (long)N * N;
  u16*   zn    = (u16*)alloc((size_t)NN * ZC * 2);
  u16*   ab    = (u16*)alloc((size_t)NN * ZC * 2);   // channel-major [c][i][k]
  u16*   bbuf  = (u16*)alloc((size_t)NN * ZC * 2);   // channel-major [c][j][k]
  u16*   xb    = (u16*)alloc((size_t)NN * ZC * 2);   // pixel-major  [i][j][c]
  float* biasz = (float*)alloc((size_t)NN * H * 4);
  u16*   snb   = (u16*)alloc((size_t)N * S * 2);
  u16*   qp    = (u16*)alloc((size_t)N * H * DP * 2);
  u16*   kp    = (u16*)alloc((size_t)N * H * DP * 2);
  u16*   vt    = (u16*)alloc((size_t)H * DP * N * 2);
  u16*   gob   = (u16*)alloc((size_t)N * S * 2);
  float* obuf  = (float*)alloc((size_t)N * S * 4);

  // transposed bf16 weights (refreshed every call; deterministic, cheap vs main work)
  enum { W_Q = 0, W_K, W_V, W_G, W_O, W_ZB, W_A, W_AG, W_B, W_BG, W_OG, W_OT,
         W_S1, W_S2, W_S3, W_Z1, W_Z2, W_Z3, NW };
  const int WIDX[NW] = { P_WQ, P_WK, P_WV, P_WG, P_WO, P_WZB, P_WA, P_WAG, P_WB, P_WBG,
                         P_WOG, P_WOT, P_WS1, P_WS2, P_WS3, P_WZ1, P_WZ2, P_WZ3 };
  const int WK[NW]  = { 384,384,384,384,384, 128, 128,128,128,128,128,128, 384,384,1536, 128,128,512 };
  const int WNC[NW] = { 384,384,384,384,384, 16,  128,128,128,128,128,128, 1536,1536,384, 512,512,128 };
  u16* wbf[2][NW];
  for (int L = 0; L < 2; ++L)
    for (int t = 0; t < NW; ++t) {
      long per = (long)WK[t] * WNC[t];
      wbf[L][t] = (u16*)alloc((size_t)per * 2);
      k_cvt_t<<<dim3(WK[t] / 16, WNC[t] / 16), 256, 0, stream>>>(PF(WIDX[t], L), wbf[L][t], WK[t], WNC[t]);
    }

  hipMemcpyAsync(s_w, s_in, (size_t)N * S * 4, hipMemcpyDeviceToDevice, stream);
  hipMemcpyAsync(z_w, z_in, (size_t)NN * ZC * 4, hipMemcpyDeviceToDevice, stream);
  // zero padded q/k/v buffers once (pad lanes are never written by the GEMMs)
  hipMemsetAsync(qp, 0, (size_t)N * H * DP * 2, stream);
  hipMemsetAsync(kp, 0, (size_t)N * H * DP * 2, stream);
  hipMemsetAsync(vt, 0, (size_t)H * DP * N * 2, stream);

  for (int L = 0; L < 2; ++L) {
    u16** wl = wbf[L];
    // zn = LN(z; tg_in, tb_in)
    k_ln<float, 128><<<(unsigned)(NN / 8), 256, 0, stream>>>(z_w, PF(P_TGIN, L), PF(P_TBIN, L), zn, NN);
    // a, b gated projections (* pair_mask), channel-major outputs
    k_dual128<1, 1><<<(unsigned)(NN / 64), 256, 0, stream>>>(zn, wl[W_AG], zn, wl[W_A], pmask, nullptr, ab, NN);
    k_dual128<1, 1><<<(unsigned)(NN / 64), 256, 0, stream>>>(zn, wl[W_BG], zn, wl[W_B], pmask, nullptr, bbuf, NN);
    // triangle einsum (reads channel-major, writes pixel-major via LDS-staged 16B stores)
    k_tri_einsum<<<dim3(N / 64, N / 32, ZC / 8), 256, 0, stream>>>(ab, bbuf, xb);
    // xn = LN(x) in place
    k_ln<u16, 128><<<(unsigned)(NN / 8), 256, 0, stream>>>(xb, PF(P_TGOUT, L), PF(P_TBOUT, L), xb, NN);
    // z += sigmoid(zn@Wog) * (xn@Wot)
    k_dual128<0, 0><<<(unsigned)(NN / 64), 256, 0, stream>>>(zn, wl[W_OG], xb, wl[W_OT], nullptr, z_w, nullptr, NN);
    // pair transition
    k_transition<128, 512><<<(unsigned)(NN / 32), 256, 0, stream>>>(
        z_w, PF(P_TZG, L), PF(P_TZB, L), wl[W_Z1], wl[W_Z2], wl[W_Z3], NN);
    // pair bias: LN(z; g_zb,b_zb) @ Wzb
    k_ln<float, 128><<<(unsigned)(NN / 8), 256, 0, stream>>>(z_w, PF(P_GZB, L), PF(P_BZB, L), zn, NN);
    k_gemm_nc16<<<(unsigned)(NN / 128), 256, 0, stream>>>(zn, wl[W_ZB], biasz, NN);
    // sn = LN(s)
    k_ln<float, 384><<<N / 8, 256, 0, stream>>>(s_w, PF(P_GS, L), PF(P_BS, L), snb, N);
    // q (+bq) / k  -> padded row-major; v -> padded transposed
    k_gemm<3, 0, 1, 0, 3><<<N / 32, 256, 0, stream>>>(snb, wl[W_Q], PF(P_BQ, L), nullptr, nullptr, qp, 384, N);
    k_gemm<3, 0, 0, 0, 3><<<N / 32, 256, 0, stream>>>(snb, wl[W_K], nullptr, nullptr, nullptr, kp, 384, N);
    k_gemm<3, 0, 0, 0, 4><<<N / 32, 256, 0, stream>>>(snb, wl[W_V], nullptr, nullptr, nullptr, vt, 384, N);
    // attention
    k_attn<<<H * (N / 16), 32, 0, stream>>>(qp, kp, vt, biasz, mask, obuf);
    // go = sigmoid(sn@Wg) * o ; s += go @ Wo
    k_gemm<3, 1, 0, 1, 1><<<N / 32, 256, 0, stream>>>(snb, wl[W_G], nullptr, obuf, nullptr, gob, 384, N);
    k_gemm<3, 0, 0, 0, 2><<<N / 32, 256, 0, stream>>>(gob, wl[W_O], nullptr, nullptr, s_w, nullptr, 384, N);
    // token transition
    k_transition<384, 1536><<<N / 32, 256, 0, stream>>>(
        s_w, PF(P_TSG, L), PF(P_TSB, L), wl[W_S1], wl[W_S2], wl[W_S3], N);
  }
}
